// ARGCNNet_57767310131498
// MI455X (gfx1250) — matrix operand order repaired and checked by reading, not compiled, not run on hardware
//
#include <hip/hip_runtime.h>
#include <hip/hip_bf16.h>
#include <math.h>

typedef __attribute__((ext_vector_type(16))) _Float16 v16h;
typedef __attribute__((ext_vector_type(8)))  float    v8f;

#define N_NODESC 50000
#define N_EDGESC 800000
#define IN_DIMC  768
#define HIDC     256
#define OUTDC    9
#define EFEATC   100
#define N_TYPESC 50
#define N_DISTC  128
#define PDROP    0.4f
#define INVKEEP  (1.0f / 0.6f)

// ---------------------------------------------------------------------------
// Gate precompute: g_t[t] = te[t,:] . gw[0:100];  g_d[d] = de[d,:] . gw[100:200]
// ---------------------------------------------------------------------------
__global__ void k_gates(const float* __restrict__ te1, const float* __restrict__ de1,
                        const float* __restrict__ g1w,
                        const float* __restrict__ te2, const float* __restrict__ de2,
                        const float* __restrict__ g2w,
                        float* __restrict__ g1t, float* __restrict__ g1d,
                        float* __restrict__ g2t, float* __restrict__ g2d) {
    int t = blockIdx.x * blockDim.x + threadIdx.x;
    if (t < N_TYPESC) {
        float s = 0.f;
        for (int f = 0; f < EFEATC; ++f) s += te1[t * EFEATC + f] * g1w[f];
        g1t[t] = s;
    } else if (t < N_TYPESC + N_DISTC) {
        int d = t - N_TYPESC;
        float s = 0.f;
        for (int f = 0; f < EFEATC; ++f) s += de1[d * EFEATC + f] * g1w[EFEATC + f];
        g1d[d] = s;
    } else if (t < 2 * N_TYPESC + N_DISTC) {
        int tt = t - (N_TYPESC + N_DISTC);
        float s = 0.f;
        for (int f = 0; f < EFEATC; ++f) s += te2[tt * EFEATC + f] * g2w[f];
        g2t[tt] = s;
    } else if (t < 2 * (N_TYPESC + N_DISTC)) {
        int d = t - (2 * N_TYPESC + N_DISTC);
        float s = 0.f;
        for (int f = 0; f < EFEATC; ++f) s += de2[d * EFEATC + f] * g2w[EFEATC + f];
        g2d[d] = s;
    }
}

// ---------------------------------------------------------------------------
// Per-edge gates: alpha = sigmoid(g_t[type] + g_d[dist] + b)
// ---------------------------------------------------------------------------
__global__ void k_alpha(const int* __restrict__ etype, const int* __restrict__ edist,
                        const float* __restrict__ g1t, const float* __restrict__ g1d,
                        const float* __restrict__ g1b,
                        const float* __restrict__ g2t, const float* __restrict__ g2d,
                        const float* __restrict__ g2b,
                        float* __restrict__ alpha1, float* __restrict__ alpha2) {
    int e = blockIdx.x * blockDim.x + threadIdx.x;
    if (e >= N_EDGESC) return;
    int t = etype[e], d = edist[e];
    float a1 = g1t[t] + g1d[d] + g1b[0];
    float a2 = g2t[t] + g2d[d] + g2b[0];
    alpha1[e] = 1.f / (1.f + expf(-a1));
    alpha2[e] = 1.f / (1.f + expf(-a2));
}

// ---------------------------------------------------------------------------
// Zero fill
// ---------------------------------------------------------------------------
__global__ void k_zero(float* __restrict__ p, size_t n) {
    size_t i = (size_t)blockIdx.x * blockDim.x + threadIdx.x;
    if (i < n) p[i] = 0.f;
}

// ---------------------------------------------------------------------------
// Pack W2 as f16, transposed [N=32][K=256]: cols 0..8 = W2_msg, 16..24 = W2_root
// ---------------------------------------------------------------------------
__global__ void k_packw2(const float* __restrict__ W2m, const float* __restrict__ W2r,
                         _Float16* __restrict__ W2hT) {
    int t = blockIdx.x * blockDim.x + threadIdx.x;   // t = n*256 + k
    if (t >= 32 * HIDC) return;
    int n = t >> 8, k = t & 255;
    float v = 0.f;
    if (n < OUTDC)                    v = W2m[k * OUTDC + n];
    else if (n >= 16 && n < 16 + OUTDC) v = W2r[k * OUTDC + (n - 16)];
    W2hT[n * HIDC + k] = (_Float16)v;
}

// ---------------------------------------------------------------------------
// GEMM1: C1[50000][512] = f16(x)[50000][768] @ f16([W1_msg | W1_root])[768][512]
// Block: 256 threads = 8 waves; block tile 32(M) x 512(N); wave tile 16 x 128.
// ---------------------------------------------------------------------------
__global__ __launch_bounds__(256) void k_gemm1(const float* __restrict__ x,
                                               const float* __restrict__ Wm,
                                               const float* __restrict__ Wr,
                                               float* __restrict__ C1) {
    __shared__ _Float16 As[32][34];     // [M][K], padded
    __shared__ _Float16 BsT[512][34];   // [N][K], padded

    const int tid  = threadIdx.x;
    const int lane = tid & 31;
    const int wave = tid >> 5;
    const int wm   = wave >> 2;         // 0..1  -> M offset 0/16
    const int wn   = wave & 3;          // 0..3  -> N offset 0/128/256/384
    const int m0   = blockIdx.x * 32;

    v8f acc[8] = {};

    for (int kk = 0; kk < IN_DIMC; kk += 32) {
        // Stage A tile (32x32 f32 -> f16)
        for (int t = tid; t < 32 * 32; t += 256) {
            int r = t >> 5, c = t & 31;
            int gr = m0 + r;
            float v = (gr < N_NODESC) ? x[(size_t)gr * IN_DIMC + kk + c] : 0.f;
            As[r][c] = (_Float16)v;
        }
        // Stage B tile transposed: BsT[n][k] = Wcat[kk+k][n]
        for (int t = tid; t < 32 * 512; t += 256) {
            int k = t >> 9, n = t & 511;
            float v = (n < HIDC) ? Wm[(size_t)(kk + k) * HIDC + n]
                                 : Wr[(size_t)(kk + k) * HIDC + (n - HIDC)];
            BsT[n][k] = (_Float16)v;
        }
        __syncthreads();

        // A fragment: 16x32 f16 (ISA 7.12.2 layout)
        v16h a;
        {
            int M  = wm * 16 + (lane & 15);
            int kb = (lane < 16) ? 0 : 8;
            const _Float16* ap = &As[M][0];
#pragma unroll
            for (int v2 = 0; v2 < 8; ++v2) {
                int K = kb + (v2 >> 2) * 16 + (v2 & 3) * 2;
                a[2 * v2]     = ap[K];
                a[2 * v2 + 1] = ap[K + 1];
            }
        }
        // 8 N-fragments of 16 columns each
#pragma unroll
        for (int f = 0; f < 8; ++f) {
            int n  = wn * 128 + f * 16 + (lane & 15);
            int kb = (lane < 16) ? 0 : 8;
            const _Float16* bp = &BsT[n][kb];
            v16h b;
#pragma unroll
            for (int e = 0; e < 16; ++e) b[e] = bp[e < 8 ? e : e + 8];
            acc[f] = __builtin_amdgcn_wmma_f32_16x16x32_f16(
                false, a, false, b, (short)0, acc[f], false, false);
        }
        __syncthreads();
    }

    // Write back: C/D layout: VGPR r -> M = r + 8*(lane>=16), N = lane%16
    {
        int Mbase = m0 + wm * 16 + ((lane >> 4) << 3);
        int nbase = wn * 128 + (lane & 15);
#pragma unroll
        for (int f = 0; f < 8; ++f) {
            int col = nbase + f * 16;
#pragma unroll
            for (int r = 0; r < 8; ++r) {
                int gr = Mbase + r;
                if (gr < N_NODESC) C1[(size_t)gr * 512 + col] = acc[f][r];
            }
        }
    }
}

// ---------------------------------------------------------------------------
// Edge pass 1: one wave per edge; each lane handles 8 of 256 features.
// agg1[dst] += alpha * C1[src][0:256]
// ---------------------------------------------------------------------------
__global__ __launch_bounds__(256) void k_edge1(const int* __restrict__ src,
                                               const int* __restrict__ dst,
                                               const float* __restrict__ alpha,
                                               const float* __restrict__ C1,
                                               float* __restrict__ agg1) {
    size_t gid = (size_t)blockIdx.x * blockDim.x + threadIdx.x;
    size_t e = gid >> 5;
    if (e >= (size_t)N_EDGESC) return;
    int p = (int)(gid & 31);
    float a = alpha[e];
    int s = src[e], d = dst[e];
    const float* xr = C1 + (size_t)s * 512 + (size_t)p * 8;   // msg half of C1
    float4 u0 = *reinterpret_cast<const float4*>(xr);
    float4 u1 = *reinterpret_cast<const float4*>(xr + 4);
    float* o = agg1 + (size_t)d * HIDC + (size_t)p * 8;
    atomicAdd(o + 0, a * u0.x); atomicAdd(o + 1, a * u0.y);
    atomicAdd(o + 2, a * u0.z); atomicAdd(o + 3, a * u0.w);
    atomicAdd(o + 4, a * u1.x); atomicAdd(o + 5, a * u1.y);
    atomicAdd(o + 6, a * u1.z); atomicAdd(o + 7, a * u1.w);
}

// ---------------------------------------------------------------------------
// Node update 1: h = relu((agg1 + root + b1) * mask1); emit f16 H for GEMM2
// ---------------------------------------------------------------------------
__global__ void k_node1(const float* __restrict__ agg1, const float* __restrict__ C1,
                        const float* __restrict__ b1, const float* __restrict__ drop1,
                        _Float16* __restrict__ Hh) {
    size_t gid = (size_t)blockIdx.x * blockDim.x + threadIdx.x;
    if (gid >= (size_t)N_NODESC * HIDC) return;
    size_t i = gid >> 8;
    int j = (int)(gid & 255);
    float m = (drop1[gid] >= PDROP) ? INVKEEP : 0.f;
    float v = agg1[gid] + C1[i * 512 + HIDC + j] + b1[j];
    v *= m;
    v = v > 0.f ? v : 0.f;
    Hh[gid] = (_Float16)v;
}

// ---------------------------------------------------------------------------
// GEMM2: C2[50000][32] = Hh[50000][256] @ W2hT^T (N padded to 32).
// Block: 256 threads = 8 waves; each wave: 16 rows x 32 cols; block: 128 rows.
// ---------------------------------------------------------------------------
__global__ __launch_bounds__(256) void k_gemm2(const _Float16* __restrict__ Hh,
                                               const _Float16* __restrict__ W2hT,
                                               float* __restrict__ C2) {
    const int tid  = threadIdx.x;
    const int lane = tid & 31;
    const int wave = tid >> 5;
    const int rbase = blockIdx.x * 128 + wave * 16;

    v8f acc[2] = {};

    for (int kk = 0; kk < HIDC; kk += 32) {
        // A fragment from global (row-major f16 H)
        v16h a;
        {
            int M  = rbase + (lane & 15);
            int Ms = (M < N_NODESC) ? M : 0;            // clamp; store is guarded
            int kb = (lane < 16) ? 0 : 8;
            const _Float16* ap = Hh + (size_t)Ms * HIDC + kk;
#pragma unroll
            for (int v2 = 0; v2 < 8; ++v2) {
                int K = kb + (v2 >> 2) * 16 + (v2 & 3) * 2;
                a[2 * v2]     = ap[K];
                a[2 * v2 + 1] = ap[K + 1];
            }
        }
#pragma unroll
        for (int f = 0; f < 2; ++f) {
            int n  = f * 16 + (lane & 15);
            int kb = (lane < 16) ? 0 : 8;
            const _Float16* bp = W2hT + (size_t)n * HIDC + kk + kb;
            v16h b;
#pragma unroll
            for (int e = 0; e < 16; ++e) b[e] = bp[e < 8 ? e : e + 8];
            acc[f] = __builtin_amdgcn_wmma_f32_16x16x32_f16(
                false, a, false, b, (short)0, acc[f], false, false);
        }
    }

    {
        int Mbase = rbase + ((lane >> 4) << 3);
        int nbase = lane & 15;
#pragma unroll
        for (int f = 0; f < 2; ++f) {
            int col = nbase + f * 16;
#pragma unroll
            for (int r = 0; r < 8; ++r) {
                int gr = Mbase + r;
                if (gr < N_NODESC) C2[(size_t)gr * 32 + col] = acc[f][r];
            }
        }
    }
}

// ---------------------------------------------------------------------------
// Edge pass 2: 16 threads per edge (9 active): agg2[dst][j] += a2 * C2[src][j]
// ---------------------------------------------------------------------------
__global__ void k_edge2(const int* __restrict__ src, const int* __restrict__ dst,
                        const float* __restrict__ alpha2, const float* __restrict__ C2,
                        float* __restrict__ agg2) {
    size_t gid = (size_t)blockIdx.x * blockDim.x + threadIdx.x;
    size_t e = gid >> 4;
    int j = (int)(gid & 15);
    if (e >= (size_t)N_EDGESC || j >= OUTDC) return;
    float a = alpha2[e];
    int s = src[e], d = dst[e];
    atomicAdd(agg2 + (size_t)d * 16 + j, a * C2[(size_t)s * 32 + j]);
}

// ---------------------------------------------------------------------------
// Node update 2: out = relu((agg2 + root + b2) * mask2)
// ---------------------------------------------------------------------------
__global__ void k_node2(const float* __restrict__ agg2, const float* __restrict__ C2,
                        const float* __restrict__ b2, const float* __restrict__ drop2,
                        float* __restrict__ out) {
    size_t gid = (size_t)blockIdx.x * blockDim.x + threadIdx.x;
    if (gid >= (size_t)N_NODESC * OUTDC) return;
    size_t i = gid / OUTDC;
    int j = (int)(gid % OUTDC);
    float m = (drop2[gid] >= PDROP) ? INVKEEP : 0.f;
    float v = agg2[i * 16 + j] + C2[i * 32 + 16 + j] + b2[j];
    v *= m;
    out[gid] = v > 0.f ? v : 0.f;
}

// ---------------------------------------------------------------------------
extern "C" void kernel_launch(void* const* d_in, const int* in_sizes, int n_in,
                              void* d_out, int out_size, void* d_ws, size_t ws_size,
                              hipStream_t stream) {
    const float* x     = (const float*)d_in[0];
    const int*   ei    = (const int*)d_in[1];
    const int*   etype = (const int*)d_in[2];
    const int*   edist = (const int*)d_in[3];
    const float* drop1 = (const float*)d_in[4];
    const float* drop2 = (const float*)d_in[5];
    const float* W1m   = (const float*)d_in[6];
    const float* W1r   = (const float*)d_in[7];
    const float* b1    = (const float*)d_in[8];
    const float* te1   = (const float*)d_in[9];
    const float* de1   = (const float*)d_in[10];
    const float* g1w   = (const float*)d_in[11];
    const float* g1b   = (const float*)d_in[12];
    const float* W2m   = (const float*)d_in[13];
    const float* W2r   = (const float*)d_in[14];
    const float* b2    = (const float*)d_in[15];
    const float* te2   = (const float*)d_in[16];
    const float* de2   = (const float*)d_in[17];
    const float* g2w   = (const float*)d_in[18];
    const float* g2b   = (const float*)d_in[19];

    const int* src = ei;
    const int* dst = ei + N_EDGESC;

    // Workspace carve-out (256B aligned)
    char* w = (char*)d_ws;
    auto carve = [&](size_t bytes) { void* p = (void*)w; w += (bytes + 255) & ~(size_t)255; return p; };
    float*    C1     = (float*)   carve((size_t)N_NODESC * 512 * 4);
    float*    agg1   = (float*)   carve((size_t)N_NODESC * HIDC * 4);
    _Float16* Hh     = (_Float16*)carve((size_t)N_NODESC * HIDC * 2);
    float*    C2     = (float*)   carve((size_t)N_NODESC * 32 * 4);
    float*    agg2   = (float*)   carve((size_t)N_NODESC * 16 * 4);
    _Float16* W2hT   = (_Float16*)carve((size_t)32 * HIDC * 2);
    float*    g1t    = (float*)   carve(N_TYPESC * 4);
    float*    g1d    = (float*)   carve(N_DISTC * 4);
    float*    g2t    = (float*)   carve(N_TYPESC * 4);
    float*    g2d    = (float*)   carve(N_DISTC * 4);
    float*    alpha1 = (float*)   carve((size_t)N_EDGESC * 4);
    float*    alpha2 = (float*)   carve((size_t)N_EDGESC * 4);

    float* out = (float*)d_out;

    // Gate scalars + per-edge alphas
    k_gates<<<2, 256, 0, stream>>>(te1, de1, g1w, te2, de2, g2w, g1t, g1d, g2t, g2d);
    k_alpha<<<(N_EDGESC + 255) / 256, 256, 0, stream>>>(etype, edist, g1t, g1d, g1b,
                                                        g2t, g2d, g2b, alpha1, alpha2);
    k_packw2<<<(32 * HIDC + 255) / 256, 256, 0, stream>>>(W2m, W2r, W2hT);

    // Layer 1
    k_zero<<<(unsigned)(((size_t)N_NODESC * HIDC + 255) / 256), 256, 0, stream>>>(
        agg1, (size_t)N_NODESC * HIDC);
    k_gemm1<<<(N_NODESC + 31) / 32, 256, 0, stream>>>(x, W1m, W1r, C1);
    k_edge1<<<(unsigned)(((size_t)N_EDGESC * 32 + 255) / 256), 256, 0, stream>>>(
        src, dst, alpha1, C1, agg1);
    k_node1<<<(unsigned)(((size_t)N_NODESC * HIDC + 255) / 256), 256, 0, stream>>>(
        agg1, C1, b1, drop1, Hh);

    // Layer 2
    k_zero<<<(unsigned)(((size_t)N_NODESC * 16 + 255) / 256), 256, 0, stream>>>(
        agg2, (size_t)N_NODESC * 16);
    k_gemm2<<<(N_NODESC + 127) / 128, 256, 0, stream>>>(Hh, W2hT, C2);
    k_edge2<<<(unsigned)(((size_t)N_EDGESC * 16 + 255) / 256), 256, 0, stream>>>(
        src, dst, alpha2, C2, agg2);
    k_node2<<<(unsigned)(((size_t)N_NODESC * OUTDC + 255) / 256), 256, 0, stream>>>(
        agg2, C2, b2, drop2, out);
}
